// GraphAttentionEmbedding_8830452760713
// MI455X (gfx1250) — matrix operand order repaired and checked by reading, not compiled, not run on hardware
//
#include <hip/hip_runtime.h>

#define NNODES 50000
#define NEDGES 250000
#define DIM    256
#define HEADS  4
#define ED_RAW 194
#define ED_PAD 224   // 194 rounded up to multiple of 32 (WMMA K step)

typedef _Float16 v16h __attribute__((ext_vector_type(16)));
typedef _Float16 h8   __attribute__((ext_vector_type(8)));
typedef float    v8f  __attribute__((ext_vector_type(8)));

union V16U { v16h v; h8 h[2]; };

// ---------------- conversion kernels ----------------

// f32 [rows x kin] -> f16 [rows x kp], zero-padded in K
__global__ void k_cvt_a(const float* __restrict__ in, _Float16* __restrict__ out,
                        int rows, int kin, int kp) {
  size_t idx = (size_t)blockIdx.x * blockDim.x + threadIdx.x;
  size_t tot = (size_t)rows * kp;
  if (idx >= tot) return;
  int c = (int)(idx % (size_t)kp);
  size_t r = idx / (size_t)kp;
  out[idx] = (c < kin) ? (_Float16)in[r * (size_t)kin + c] : (_Float16)0.0f;
}

// W f32 [kin x 256] -> Wt f16 [256 x kp] (transposed, zero-padded)
__global__ void k_cvt_wt(const float* __restrict__ W, _Float16* __restrict__ Wt,
                         int kin, int kp) {
  size_t idx = (size_t)blockIdx.x * blockDim.x + threadIdx.x;
  size_t tot = (size_t)DIM * kp;
  if (idx >= tot) return;
  int kpos = (int)(idx % (size_t)kp);
  int n    = (int)(idx / (size_t)kp);
  Wt[idx] = (kpos < kin) ? (_Float16)W[(size_t)kpos * DIM + n] : (_Float16)0.0f;
}

// ---------------- WMMA GEMM ----------------
// C[M x 256] = A16[M x Kp] * Bt[256 x Kp]^T (+ bias)
// block = 256 threads = 8 waves; block covers 16 rows x 256 cols;
// wave w handles column tiles [32w, 32w+16) and [32w+16, 32w+32).
__global__ void k_wmma_gemm(const _Float16* __restrict__ A,
                            const _Float16* __restrict__ Bt,
                            const float* __restrict__ bias,
                            float* __restrict__ C,
                            int Kp) {
  const int lane = threadIdx.x & 31;
  const int wave = threadIdx.x >> 5;
  const int hi   = lane >> 4;      // lane group 0/1
  const int l16  = lane & 15;
  const int mrow = blockIdx.x * 16 + l16;
  const int n0   = wave * 32;

  const _Float16* arow  = A  + (size_t)mrow * Kp;
  const _Float16* brow0 = Bt + (size_t)(n0 + l16)      * Kp;
  const _Float16* brow1 = Bt + (size_t)(n0 + 16 + l16) * Kp;

  v8f acc0 = {}; v8f acc1 = {};
  for (int kk = 0; kk < Kp; kk += 32) {
    // A frag (16x32 f16): lane group g holds 16 consecutive K at kk + 16g for its row
    V16U a, b0, b1;
    const _Float16* ap = arow + kk + hi * 16;
    a.h[0] = *(const h8*)(ap);
    a.h[1] = *(const h8*)(ap + 8);
    // B frag (32x16 f16): lane group g holds 16 consecutive K at kk + 16g for its column
    const _Float16* bp0 = brow0 + kk + hi * 16;
    b0.h[0] = *(const h8*)(bp0);
    b0.h[1] = *(const h8*)(bp0 + 8);
    const _Float16* bp1 = brow1 + kk + hi * 16;
    b1.h[0] = *(const h8*)(bp1);
    b1.h[1] = *(const h8*)(bp1 + 8);

    acc0 = __builtin_amdgcn_wmma_f32_16x16x32_f16(false, a.v, false, b0.v,
                                                  (short)0, acc0, false, false);
    acc1 = __builtin_amdgcn_wmma_f32_16x16x32_f16(false, a.v, false, b1.v,
                                                  (short)0, acc1, false, false);
  }

  // C/D layout: lane = col (l16), VGPR r = row r + 8*hi
  const int orow0 = blockIdx.x * 16 + hi * 8;
  const int c0 = n0 + l16;
  const float bb0 = bias ? bias[c0]      : 0.0f;
  const float bb1 = bias ? bias[c0 + 16] : 0.0f;
#pragma unroll
  for (int r = 0; r < 8; ++r) {
    size_t ro = (size_t)(orow0 + r) * DIM;
    C[ro + c0]      = acc0[r] + bb0;
    C[ro + c0 + 16] = acc1[r] + bb1;
  }
}

// ---------------- segment softmax helpers ----------------

__device__ __forceinline__ unsigned ordf(float f) {
  unsigned u = __float_as_uint(f);
  return (u & 0x80000000u) ? ~u : (u | 0x80000000u);
}
__device__ __forceinline__ float unordf(unsigned u) {
  return (u & 0x80000000u) ? __uint_as_float(u & 0x7fffffffu) : __uint_as_float(~u);
}

__global__ void k_init_seg(unsigned* __restrict__ maxv, float* __restrict__ denom, int n) {
  int idx = blockIdx.x * blockDim.x + threadIdx.x;
  if (idx < n) { maxv[idx] = 0u; denom[idx] = 0.0f; }
}

// wave per edge: lane L covers channels [8L, 8L+8); head = L/8
__global__ void k_edge_logits(const int* __restrict__ src, const int* __restrict__ dst,
                              const float* __restrict__ q, const float* __restrict__ k,
                              const float* __restrict__ e,
                              float* __restrict__ logits, unsigned* __restrict__ maxv) {
  int edge = blockIdx.x * 8 + (threadIdx.x >> 5);
  if (edge >= NEDGES) return;
  int lane = threadIdx.x & 31;
  int s = src[edge], d = dst[edge];
  int off = lane * 8;
  const float* qp = q + (size_t)d * DIM + off;
  const float* kp = k + (size_t)s * DIM + off;
  const float* ep = e + (size_t)edge * DIM + off;
  float acc = 0.0f;
#pragma unroll
  for (int i = 0; i < 8; ++i) acc += qp[i] * (kp[i] + ep[i]);
  acc += __shfl_down(acc, 4);
  acc += __shfl_down(acc, 2);
  acc += __shfl_down(acc, 1);
  if ((lane & 7) == 0) {
    int h = lane >> 3;
    float lg = acc * 0.125f;  // 1/sqrt(64)
    logits[(size_t)edge * HEADS + h] = lg;
    atomicMax(&maxv[(size_t)d * HEADS + h], ordf(lg));
  }
}

__global__ void k_edge_exp(const int* __restrict__ dst, float* __restrict__ logits,
                           const unsigned* __restrict__ maxv, float* __restrict__ denom) {
  size_t idx = (size_t)blockIdx.x * blockDim.x + threadIdx.x;
  if (idx >= (size_t)NEDGES * HEADS) return;
  int h = (int)(idx & 3);
  int d = dst[idx >> 2];
  float m  = unordf(maxv[(size_t)d * HEADS + h]);
  float ex = __expf(logits[idx] - m);
  logits[idx] = ex;  // reuse buffer as exp(logit - max)
  atomicAdd(&denom[(size_t)d * HEADS + h], ex);
}

// wave per edge: scatter (v[src]+e)*alpha into out[dst]
__global__ void k_edge_scatter(const int* __restrict__ src, const int* __restrict__ dst,
                               const float* __restrict__ v, const float* __restrict__ e,
                               const float* __restrict__ ex, const float* __restrict__ denom,
                               float* __restrict__ out) {
  int edge = blockIdx.x * 8 + (threadIdx.x >> 5);
  if (edge >= NEDGES) return;
  int lane = threadIdx.x & 31;
  int s = src[edge], d = dst[edge];
  int h = lane >> 3;
  int off = lane * 8;
  float a = ex[(size_t)edge * HEADS + h] / (denom[(size_t)d * HEADS + h] + 1e-16f);
  const float* vp = v + (size_t)s * DIM + off;
  const float* ep = e + (size_t)edge * DIM + off;
  float* op = out + (size_t)d * DIM + off;
#pragma unroll
  for (int i = 0; i < 8; ++i) atomicAdd(&op[i], (vp[i] + ep[i]) * a);
}

__global__ void k_relu(float* __restrict__ h, size_t n) {
  size_t idx = (size_t)blockIdx.x * blockDim.x + threadIdx.x;
  if (idx < n) h[idx] = fmaxf(h[idx], 0.0f);
}

// ---------------- orchestration ----------------

extern "C" void kernel_launch(void* const* d_in, const int* in_sizes, int n_in,
                              void* d_out, int out_size, void* d_ws, size_t ws_size,
                              hipStream_t stream) {
  (void)in_sizes; (void)n_in; (void)out_size; (void)ws_size;

  const float* x    = (const float*)d_in[0];
  const int*   eidx = (const int*)d_in[1];
  const float* ea   = (const float*)d_in[2];
  const int* src = eidx;
  const int* dst = eidx + NEDGES;

  // workspace carve-out (256B aligned)
  char* ws = (char*)d_ws;
  size_t off = 0;
  auto take = [&](size_t b) -> char* {
    char* p = ws + off;
    off = (off + b + 255) & ~(size_t)255;
    return p;
  };
  float*     qbuf   = (float*)take((size_t)NNODES * DIM * 4);
  float*     kbuf   = (float*)take((size_t)NNODES * DIM * 4);
  float*     vbuf   = (float*)take((size_t)NNODES * DIM * 4);
  float*     ebuf   = (float*)take((size_t)NEDGES * DIM * 4);
  float*     hbuf   = (float*)take((size_t)NNODES * DIM * 4);
  float*     logits = (float*)take((size_t)NEDGES * HEADS * 4);
  unsigned*  maxv   = (unsigned*)take((size_t)NNODES * HEADS * 4);
  float*     denom  = (float*)take((size_t)NNODES * HEADS * 4);
  _Float16*  a16    = (_Float16*)take((size_t)NNODES * DIM * 2);
  _Float16*  ea16   = (_Float16*)take((size_t)NEDGES * ED_PAD * 2);
  _Float16*  wt16   = (_Float16*)take((size_t)DIM * DIM * 2);

  dim3 blk(256);
  auto cdiv = [](size_t a, size_t b) { return (unsigned)((a + b - 1) / b); };

  // edge_attr -> f16 padded (shared by both layers)
  k_cvt_a<<<cdiv((size_t)NEDGES * ED_PAD, 256), blk, 0, stream>>>(ea, ea16, NEDGES, ED_RAW, ED_PAD);

  const float* layer_in = x;
  for (int layer = 0; layer < 2; ++layer) {
    const int pb = (layer == 0) ? 3 : 12;
    const float* Wq = (const float*)d_in[pb + 0];
    const float* bq = (const float*)d_in[pb + 1];
    const float* Wk = (const float*)d_in[pb + 2];
    const float* bk = (const float*)d_in[pb + 3];
    const float* Wv = (const float*)d_in[pb + 4];
    const float* bv = (const float*)d_in[pb + 5];
    const float* We = (const float*)d_in[pb + 6];
    const float* Wsk = (const float*)d_in[pb + 7];
    const float* bsk = (const float*)d_in[pb + 8];
    float* outbuf = (layer == 0) ? hbuf : (float*)d_out;

    // activations -> f16
    k_cvt_a<<<cdiv((size_t)NNODES * DIM, 256), blk, 0, stream>>>(layer_in, a16, NNODES, DIM, DIM);

    // q, k, v projections (WMMA)
    k_cvt_wt<<<cdiv((size_t)DIM * DIM, 256), blk, 0, stream>>>(Wq, wt16, DIM, DIM);
    k_wmma_gemm<<<NNODES / 16, blk, 0, stream>>>(a16, wt16, bq, qbuf, DIM);
    k_cvt_wt<<<cdiv((size_t)DIM * DIM, 256), blk, 0, stream>>>(Wk, wt16, DIM, DIM);
    k_wmma_gemm<<<NNODES / 16, blk, 0, stream>>>(a16, wt16, bk, kbuf, DIM);
    k_cvt_wt<<<cdiv((size_t)DIM * DIM, 256), blk, 0, stream>>>(Wv, wt16, DIM, DIM);
    k_wmma_gemm<<<NNODES / 16, blk, 0, stream>>>(a16, wt16, bv, vbuf, DIM);

    // edge embedding e = edge_attr @ We (WMMA, no bias)
    k_cvt_wt<<<cdiv((size_t)DIM * ED_PAD, 256), blk, 0, stream>>>(We, wt16, ED_RAW, ED_PAD);
    k_wmma_gemm<<<NEDGES / 16, blk, 0, stream>>>(ea16, wt16, nullptr, ebuf, ED_PAD);

    // skip connection directly into the output buffer
    k_cvt_wt<<<cdiv((size_t)DIM * DIM, 256), blk, 0, stream>>>(Wsk, wt16, DIM, DIM);
    k_wmma_gemm<<<NNODES / 16, blk, 0, stream>>>(a16, wt16, bsk, outbuf, DIM);

    // segment softmax + scatter aggregation
    k_init_seg<<<cdiv((size_t)NNODES * HEADS, 256), blk, 0, stream>>>(maxv, denom, NNODES * HEADS);
    k_edge_logits<<<cdiv(NEDGES, 8), blk, 0, stream>>>(src, dst, qbuf, kbuf, ebuf, logits, maxv);
    k_edge_exp<<<cdiv((size_t)NEDGES * HEADS, 256), blk, 0, stream>>>(dst, logits, maxv, denom);
    k_edge_scatter<<<cdiv(NEDGES, 8), blk, 0, stream>>>(src, dst, vbuf, ebuf, logits, denom, outbuf);

    if (layer == 0) {
      k_relu<<<cdiv((size_t)NNODES * DIM, 256), blk, 0, stream>>>(hbuf, (size_t)NNODES * DIM);
      layer_in = hbuf;
    }
  }
}
